// TransformerPredictor_48610439856616
// MI455X (gfx1250) — compile-verified, matrix operand order checked by reading, not run on hardware
//
#include <hip/hip_runtime.h>

// ---------------------------------------------------------------------------
// Types
// ---------------------------------------------------------------------------
typedef __attribute__((ext_vector_type(16))) __bf16          bf16x16;
typedef __attribute__((ext_vector_type(8)))  __bf16          bf16x8;
typedef __attribute__((ext_vector_type(8)))  float           v8f;
typedef __attribute__((ext_vector_type(8)))  unsigned short  u16x8;

// Problem constants (match the reference)
#define B_    32
#define S_    256
#define T2_   512          // 2S
#define SD_   512
#define AD_   256
#define E_    1024
#define F_    2048
#define H_    16
#define D_    64
#define MROWS 16384        // B * 2S

// ---------------------------------------------------------------------------
// Helpers
// ---------------------------------------------------------------------------
__device__ __forceinline__ unsigned short f2bf(float f) {
    unsigned u = __builtin_bit_cast(unsigned, f);
    u += 0x7FFFu + ((u >> 16) & 1u);          // round-to-nearest-even
    return (unsigned short)(u >> 16);
}
__device__ __forceinline__ float bf2f(unsigned short h) {
    return __builtin_bit_cast(float, (unsigned)h << 16);
}
__device__ __forceinline__ v8f zero8() {
    v8f z;
    #pragma unroll
    for (int i = 0; i < 8; ++i) z[i] = 0.0f;
    return z;
}

// Async global -> LDS copy of 16 bytes per lane (CDNA5, tracked by ASYNCcnt).
// LDS generic pointers carry the workgroup-relative byte offset in bits [31:0].
__device__ __forceinline__ void async_copy16(const unsigned short* lds,
                                             const unsigned short* g) {
    unsigned lo = (unsigned)(unsigned long long)(uintptr_t)lds;
    unsigned long long ga = (unsigned long long)(uintptr_t)g;
    asm volatile("global_load_async_to_lds_b128 %0, %1, off"
                 :: "v"(lo), "v"(ga) : "memory");
}
__device__ __forceinline__ void wait_async0() {
    asm volatile("s_wait_asynccnt 0x0" ::: "memory");
}

// Load a 16x32 bf16 A/B fragment from an LDS tile laid out row-major
// (stride in halves, 16B-aligned rows).  CDNA5 16-bit operand layout:
// lane (l&15) = row; halves 0..7 -> k = klo..klo+7, halves 8..15 -> k = klo+16..,
// klo = (l>>4)*8.  Two 16-byte chunks per lane.
__device__ __forceinline__ bf16x16 ld_frag(const unsigned short* p, int row0,
                                           int stride, int kOff) {
    int lane = threadIdx.x & 31;
    const unsigned short* r = p + (size_t)(row0 + (lane & 15)) * stride
                                + kOff + ((lane >> 4) << 3);
    bf16x8 lo = __builtin_bit_cast(bf16x8, *(const u16x8*)r);
    bf16x8 hi = __builtin_bit_cast(bf16x8, *(const u16x8*)(r + 16));
    return __builtin_shufflevector(lo, hi, 0,1,2,3,4,5,6,7,8,9,10,11,12,13,14,15);
}

__device__ __forceinline__ v8f wmma_bf16(bf16x16 a, bf16x16 b, v8f c) {
    return __builtin_amdgcn_wmma_f32_16x16x32_bf16(false, a, false, b,
                                                   (short)0, c, false, false);
}

// ---------------------------------------------------------------------------
// Tiled WMMA GEMM:  C(M,N) = A(M,K) * W(N,K)^T + bias(N)
// A and W are bf16 (row strides lda/ldb elements).  Double-buffered LDS tiles
// filled with GLOBAL_LOAD_ASYNC_TO_LDS_B128.  BM=128, BN=256, BK=32,
// 256 threads (8 waves) in a 2x4 wave grid; each wave computes a 64x64 tile
// = 16 wmma per K-step against 16 ds_load_b128 (1.0 DS load per WMMA).
// ---------------------------------------------------------------------------
#define BM  128
#define BN  256
#define BK  32
#define LDT 40   // BK + 8 halves padding (80 bytes, 16B aligned)

template <bool RELU, bool OBF>
__global__ __launch_bounds__(256)
void gemm_wmma(const unsigned short* __restrict__ A, long lda,
               const unsigned short* __restrict__ Bw, long ldb,
               const float* __restrict__ bias,
               void* __restrict__ Cv, long ldc,
               int M, int N, int K) {
    __shared__ __align__(16) unsigned short lA[2][BM * LDT];
    __shared__ __align__(16) unsigned short lB[2][BN * LDT];

    const int tid  = threadIdx.x;
    const int lane = tid & 31;
    const int wid  = tid >> 5;
    const int wm   = wid & 1;        // 2 wave rows (64 rows each)
    const int wn   = wid >> 1;       // 4 wave cols (64 cols each)
    const int bm0  = blockIdx.y * BM;
    const int bn0  = blockIdx.x * BN;

    // Stage one BK-slice (A tile + B tile) into LDS buffer `buf` via async DMA.
    auto stage = [&](int buf, int k0) {
        #pragma unroll
        for (int cc = 0; cc < 6; ++cc) {
            int c = tid + cc * 256;         // 0..1535: 512 A chunks + 1024 B
            if (c < BM * 4) {
                int row = c >> 2, ck = (c & 3) * 8;
                int gm = bm0 + row;
                if (gm < M) async_copy16(&lA[buf][row * LDT + ck],
                                         A + (size_t)gm * lda + k0 + ck);
            } else {
                int cb  = c - BM * 4;
                int row = cb >> 2, ck = (cb & 3) * 8;
                int gn = bn0 + row;
                if (gn < N) async_copy16(&lB[buf][row * LDT + ck],
                                         Bw + (size_t)gn * ldb + k0 + ck);
            }
        }
    };

    v8f acc[4][4];
    #pragma unroll
    for (int i = 0; i < 4; ++i)
        #pragma unroll
        for (int j = 0; j < 4; ++j) acc[i][j] = zero8();

    stage(0, 0);
    wait_async0();
    __syncthreads();

    int buf = 0;
    for (int k0 = 0; k0 < K; k0 += BK) {
        if (k0 + BK < K) stage(buf ^ 1, k0 + BK);   // overlap with WMMA below

        bf16x16 af[4], bfr[4];
        #pragma unroll
        for (int i = 0; i < 4; ++i) af[i]  = ld_frag(lA[buf], wm * 64 + i * 16, LDT, 0);
        #pragma unroll
        for (int j = 0; j < 4; ++j) bfr[j] = ld_frag(lB[buf], wn * 64 + j * 16, LDT, 0);
        #pragma unroll
        for (int i = 0; i < 4; ++i)
            #pragma unroll
            for (int j = 0; j < 4; ++j)
                acc[i][j] = wmma_bf16(af[i], bfr[j], acc[i][j]);

        wait_async0();
        __syncthreads();
        buf ^= 1;
    }

    // Epilogue: C frag layout m = r + 8*(lane>>4), n = lane&15
    #pragma unroll
    for (int i = 0; i < 4; ++i)
        #pragma unroll
        for (int j = 0; j < 4; ++j) {
            int gn = bn0 + wn * 64 + j * 16 + (lane & 15);
            int mb = bm0 + wm * 64 + i * 16 + 8 * (lane >> 4);
            if (gn >= N) continue;
            float bv = bias[gn];
            #pragma unroll
            for (int r = 0; r < 8; ++r) {
                int gm = mb + r;
                if (gm < M) {
                    float v = acc[i][j][r] + bv;
                    if (RELU) v = fmaxf(v, 0.0f);
                    if (OBF) ((unsigned short*)Cv)[(size_t)gm * ldc + gn] = f2bf(v);
                    else     ((float*)Cv)[(size_t)gm * ldc + gn] = v;
                }
            }
        }
}

// ---------------------------------------------------------------------------
// Encoder self-attention: one wave per (t, h).  Sequence axis is B=32 (the
// reference attends over axis 0), head dim 64.  QK^T and attn*V via WMMA.
// qkv: bf16 (MROWS, 3072) with q|k|v at col offsets 0|1024|2048.
// ---------------------------------------------------------------------------
__global__ __launch_bounds__(32)
void attn_enc_kernel(const unsigned short* __restrict__ qkv,
                     unsigned short* __restrict__ aout) {
    const int t = blockIdx.x, h = blockIdx.y, lane = threadIdx.x;

    __shared__ __align__(16) unsigned short lq[32 * 72];
    __shared__ __align__(16) unsigned short lk[32 * 72];
    __shared__ __align__(16) unsigned short lvt[64 * 40];  // V transposed (d, b)
    __shared__ __align__(16) float          lsc[32 * 36];
    __shared__ __align__(16) unsigned short lat[32 * 40];

    const size_t rbase = ((size_t)lane * T2_ + t) * 3072 + (size_t)h * D_;
    #pragma unroll
    for (int c = 0; c < 8; ++c) {
        u16x8 vq = *(const u16x8*)(qkv + rbase + c * 8);
        u16x8 vk = *(const u16x8*)(qkv + rbase + 1024 + c * 8);
        u16x8 vv = *(const u16x8*)(qkv + rbase + 2048 + c * 8);
        *(u16x8*)&lq[lane * 72 + c * 8] = vq;
        *(u16x8*)&lk[lane * 72 + c * 8] = vk;
        #pragma unroll
        for (int z = 0; z < 8; ++z) lvt[(c * 8 + z) * 40 + lane] = vv[z];
    }
    __syncthreads();

    // scores(32x32) = Q(32x64) * K(32x64)^T, scaled by 1/sqrt(64)
    #pragma unroll
    for (int ti = 0; ti < 2; ++ti)
        #pragma unroll
        for (int tj = 0; tj < 2; ++tj) {
            v8f acc = zero8();
            #pragma unroll
            for (int kk = 0; kk < 64; kk += 32) {
                bf16x16 a = ld_frag(lq, ti * 16, 72, kk);
                bf16x16 b = ld_frag(lk, tj * 16, 72, kk);
                acc = wmma_bf16(a, b, acc);
            }
            int n  = tj * 16 + (lane & 15);
            int mb = ti * 16 + 8 * (lane >> 4);
            #pragma unroll
            for (int r = 0; r < 8; ++r) lsc[(mb + r) * 36 + n] = acc[r] * 0.125f;
        }
    __syncthreads();

    // row-wise softmax (lane i owns query row i)
    {
        float mx = -1e30f;
        for (int j = 0; j < 32; ++j) mx = fmaxf(mx, lsc[lane * 36 + j]);
        float sum = 0.0f;
        for (int j = 0; j < 32; ++j) {
            float ev = __expf(lsc[lane * 36 + j] - mx);
            lsc[lane * 36 + j] = ev;
            sum += ev;
        }
        float inv = 1.0f / sum;
        for (int j = 0; j < 32; ++j) lat[lane * 40 + j] = f2bf(lsc[lane * 36 + j] * inv);
    }
    __syncthreads();

    // out(32x64) = attn(32x32) * V(32x64)
    #pragma unroll
    for (int ti = 0; ti < 2; ++ti)
        #pragma unroll
        for (int tn = 0; tn < 4; ++tn) {
            v8f acc = zero8();
            bf16x16 a = ld_frag(lat, ti * 16, 40, 0);
            bf16x16 b = ld_frag(lvt, tn * 16, 40, 0);
            acc = wmma_bf16(a, b, acc);
            int d  = tn * 16 + (lane & 15);
            int mb = ti * 16 + 8 * (lane >> 4);
            #pragma unroll
            for (int r = 0; r < 8; ++r) {
                int bb = mb + r;
                aout[((size_t)bb * T2_ + t) * E_ + (size_t)h * D_ + d] = f2bf(acc[r]);
            }
        }
}

// ---------------------------------------------------------------------------
// Forward-dynamics head attention: 1 query, 2 keys, per (t, b, h).  One wave.
// ---------------------------------------------------------------------------
__global__ __launch_bounds__(32)
void fdh_attn_kernel(const unsigned short* __restrict__ qkv,
                     unsigned short* __restrict__ out) {
    const int t = blockIdx.x + 1, b = blockIdx.y, h = blockIdx.z;
    const int lane = threadIdx.x;
    const size_t rq  = ((size_t)b * T2_ + 2 * t) * 3072 + (size_t)h * D_;
    const size_t rk0 = ((size_t)b * T2_ + 2 * t - 2) * 3072 + 1024 + (size_t)h * D_;
    const size_t rk1 = rk0 + 3072;
    const size_t rv0 = rk0 + 1024;
    const size_t rv1 = rv0 + 3072;

    float q0 = bf2f(qkv[rq + lane]), q1 = bf2f(qkv[rq + lane + 32]);
    float s0 = q0 * bf2f(qkv[rk0 + lane]) + q1 * bf2f(qkv[rk0 + lane + 32]);
    float s1 = q0 * bf2f(qkv[rk1 + lane]) + q1 * bf2f(qkv[rk1 + lane + 32]);
    #pragma unroll
    for (int m = 16; m >= 1; m >>= 1) {
        s0 += __shfl_xor(s0, m, 32);
        s1 += __shfl_xor(s1, m, 32);
    }
    s0 *= 0.125f; s1 *= 0.125f;
    float mx = fmaxf(s0, s1);
    float e0 = __expf(s0 - mx), e1 = __expf(s1 - mx);
    float inv = 1.0f / (e0 + e1);
    float w0 = e0 * inv, w1 = e1 * inv;

    const size_t orow = ((size_t)(t - 1) * B_ + b) * E_ + (size_t)h * D_;
    out[orow + lane]      = f2bf(w0 * bf2f(qkv[rv0 + lane])      + w1 * bf2f(qkv[rv1 + lane]));
    out[orow + lane + 32] = f2bf(w0 * bf2f(qkv[rv0 + lane + 32]) + w1 * bf2f(qkv[rv1 + lane + 32]));
}

// ---------------------------------------------------------------------------
// Masked-action head attention: per (i, b, h); U keys (U <= 32).  One wave.
// ---------------------------------------------------------------------------
__global__ __launch_bounds__(32)
void mah_attn_kernel(const unsigned short* __restrict__ qkv,
                     const int* __restrict__ mia,
                     const int* __restrict__ um, int U,
                     unsigned short* __restrict__ out) {
    const int i = blockIdx.x, b = blockIdx.y, h = blockIdx.z;
    const int lane = threadIdx.x;
    __shared__ float qs[64];
    __shared__ float w[32];
    __shared__ int   uml[32];

    const int p = 2 * mia[i] + 1;
    const size_t qrow = ((size_t)b * T2_ + p) * 3072 + (size_t)h * D_;
    qs[lane]      = bf2f(qkv[qrow + lane]);
    qs[lane + 32] = bf2f(qkv[qrow + lane + 32]);
    if (lane < U) uml[lane] = um[lane];
    __syncthreads();

    float s = -1e30f;
    if (lane < U) {
        const size_t krow = ((size_t)b * T2_ + uml[lane]) * 3072 + 1024 + (size_t)h * D_;
        float acc = 0.0f;
        for (int d = 0; d < 64; ++d) acc += qs[d] * bf2f(qkv[krow + d]);
        s = acc * 0.125f;
    }
    float mx = s;
    #pragma unroll
    for (int m = 16; m >= 1; m >>= 1) mx = fmaxf(mx, __shfl_xor(mx, m, 32));
    float e = (lane < U) ? __expf(s - mx) : 0.0f;
    float sum = e;
    #pragma unroll
    for (int m = 16; m >= 1; m >>= 1) sum += __shfl_xor(sum, m, 32);
    w[lane] = e / sum;
    __syncthreads();

    float o0 = 0.0f, o1 = 0.0f;
    for (int j = 0; j < U; ++j) {
        const size_t vrow = ((size_t)b * T2_ + uml[j]) * 3072 + 2048 + (size_t)h * D_;
        float wj = w[j];
        o0 += wj * bf2f(qkv[vrow + lane]);
        o1 += wj * bf2f(qkv[vrow + lane + 32]);
    }
    const size_t orow = ((size_t)i * B_ + b) * E_ + (size_t)h * D_;
    out[orow + lane]      = f2bf(o0);
    out[orow + lane + 32] = f2bf(o1);
}

// ---------------------------------------------------------------------------
// Residual + LayerNorm over E=1024: LN(res + y)*g + bb -> f32 out + bf16 out
// ---------------------------------------------------------------------------
__global__ __launch_bounds__(256)
void ln_kernel(const float* __restrict__ res, const float* __restrict__ y,
               const float* __restrict__ g, const float* __restrict__ bb,
               float* __restrict__ outf, unsigned short* __restrict__ outb) {
    const int row = blockIdx.x;
    const float* r  = res + (size_t)row * E_;
    const float* yy = y   + (size_t)row * E_;
    float v[4], s = 0.0f, s2 = 0.0f;
    #pragma unroll
    for (int j = 0; j < 4; ++j) {
        int e = threadIdx.x + j * 256;
        float tv = r[e] + yy[e];
        v[j] = tv; s += tv; s2 += tv * tv;
    }
    #pragma unroll
    for (int m = 16; m >= 1; m >>= 1) {
        s  += __shfl_xor(s, m, 32);
        s2 += __shfl_xor(s2, m, 32);
    }
    __shared__ float ss[8], ss2[8], mv[2];
    int wid = threadIdx.x >> 5;
    if ((threadIdx.x & 31) == 0) { ss[wid] = s; ss2[wid] = s2; }
    __syncthreads();
    if (threadIdx.x == 0) {
        float S = 0, S2 = 0;
        for (int k = 0; k < 8; ++k) { S += ss[k]; S2 += ss2[k]; }
        float mean = S * (1.0f / E_);
        float var  = S2 * (1.0f / E_) - mean * mean;
        mv[0] = mean; mv[1] = rsqrtf(var + 1e-5f);
    }
    __syncthreads();
    float mean = mv[0], inv = mv[1];
    float*          of = outf + (size_t)row * E_;
    unsigned short* ob = outb + (size_t)row * E_;
    #pragma unroll
    for (int j = 0; j < 4; ++j) {
        int e = threadIdx.x + j * 256;
        float o = (v[j] - mean) * inv * g[e] + bb[e];
        of[e] = o;
        ob[e] = f2bf(o);
    }
}

// ---------------------------------------------------------------------------
// Utilities
// ---------------------------------------------------------------------------
__global__ void cvt_kernel(const float* __restrict__ s,
                           unsigned short* __restrict__ d, size_t n) {
    size_t i = (size_t)blockIdx.x * 256 + threadIdx.x;
    if (i < n) d[i] = f2bf(s[i]);
}

// es/ea (f32) -> interleaved x (f32) + xbf/ebf (bf16)
__global__ void interleave_kernel(const float* __restrict__ es,
                                  const float* __restrict__ ea,
                                  float* __restrict__ x,
                                  unsigned short* __restrict__ xbf,
                                  unsigned short* __restrict__ ebf) {
    size_t i = (size_t)blockIdx.x * 256 + threadIdx.x;
    const size_t n = (size_t)B_ * S_ * E_;
    if (i >= n) return;
    int e = (int)(i % E_);
    size_t bs = i / E_;
    int s = (int)(bs % S_);
    int b = (int)(bs / S_);
    size_t r0 = ((size_t)b * T2_ + 2 * s) * E_ + e;
    size_t r1 = r0 + E_;
    float v0 = es[i], v1 = ea[i];
    unsigned short h0 = f2bf(v0), h1 = f2bf(v1);
    x[r0] = v0;   x[r1] = v1;
    xbf[r0] = h0; xbf[r1] = h1;
    ebf[r0] = h0; ebf[r1] = h1;
}

__global__ void fillz_kernel(float* __restrict__ p, size_t n) {
    size_t i = (size_t)blockIdx.x * 256 + threadIdx.x;
    if (i < n) p[i] = 0.0f;
}

__global__ void scatter_masked_kernel(const float* __restrict__ tmp,
                                      float* __restrict__ o, int Ma) {
    size_t i = (size_t)blockIdx.x * 256 + threadIdx.x;
    size_t n = (size_t)Ma * B_ * AD_;
    if (i >= n) return;
    int c = (int)(i % AD_);
    size_t rb = i / AD_;
    int b = (int)(rb & 31);
    int ii = (int)(rb >> 5);
    o[(size_t)b * Ma * AD_ + (size_t)ii * AD_ + c] = tmp[i];
}

__global__ void idxcopy_kernel(const int* __restrict__ s, int n,
                               float* __restrict__ o) {
    int i = blockIdx.x * 256 + threadIdx.x;
    if (i < n) o[i] = (float)s[i];
}

// ---------------------------------------------------------------------------
// Host-side: convert weight f32->bf16 into staging, then run WMMA GEMM
// ---------------------------------------------------------------------------
static inline void launch_gemm(hipStream_t st,
                               const unsigned short* A, long lda,
                               const float* W, unsigned short* WBF,
                               const float* bias, void* C, bool obf, long ldc,
                               int M, int N, int K, bool relu) {
    size_t wn = (size_t)N * K;
    cvt_kernel<<<(unsigned)((wn + 255) / 256), 256, 0, st>>>(W, WBF, wn);
    dim3 g((N + BN - 1) / BN, (M + BM - 1) / BM), blk(256);
    if (obf) {
        if (relu) gemm_wmma<true,  true ><<<g, blk, 0, st>>>(A, lda, WBF, K, bias, C, ldc, M, N, K);
        else      gemm_wmma<false, true ><<<g, blk, 0, st>>>(A, lda, WBF, K, bias, C, ldc, M, N, K);
    } else {
        if (relu) gemm_wmma<true,  false><<<g, blk, 0, st>>>(A, lda, WBF, K, bias, C, ldc, M, N, K);
        else      gemm_wmma<false, false><<<g, blk, 0, st>>>(A, lda, WBF, K, bias, C, ldc, M, N, K);
    }
}

// ---------------------------------------------------------------------------
// Orchestration
// ---------------------------------------------------------------------------
extern "C" void kernel_launch(void* const* d_in, const int* in_sizes, int n_in,
                              void* d_out, int out_size, void* d_ws, size_t ws_size,
                              hipStream_t stream) {
    (void)n_in; (void)out_size; (void)ws_size;

    const float* states  = (const float*)d_in[0];
    const float* actions = (const float*)d_in[1];
    const int*   mis     = (const int*)d_in[2];  const int nMis = in_sizes[2];
    const int*   mia     = (const int*)d_in[3];  const int Ma   = in_sizes[3];
    const int*   umi     = (const int*)d_in[4];  const int U    = in_sizes[4];
    const float* Ws  = (const float*)d_in[5];
    const float* bs_ = (const float*)d_in[6];
    const float* Wa  = (const float*)d_in[7];
    const float* ba_ = (const float*)d_in[8];
    const float* outW = (const float*)d_in[57];
    const float* outB = (const float*)d_in[58];
    const float* fWqkv = (const float*)d_in[59]; const float* fbqkv = (const float*)d_in[60];
    const float* fWo   = (const float*)d_in[61]; const float* fbo   = (const float*)d_in[62];
    const float* fWout = (const float*)d_in[63]; const float* fbout = (const float*)d_in[64];
    const float* mWqkv = (const float*)d_in[65]; const float* mbqkv = (const float*)d_in[66];
    const float* mWo   = (const float*)d_in[67]; const float* mbo   = (const float*)d_in[68];
    const float* mWout = (const float*)d_in[69]; const float* mbout = (const float*)d_in[70];

    // Workspace layout (everything bf16 except LN residual stream + f32 temp)
    float*          X    = (float*)d_ws;                         // 16384x1024 f32
    float*          TMP  = X + (size_t)MROWS * E_;               // 16384x1024 f32
    unsigned short* QKV  = (unsigned short*)(TMP + (size_t)MROWS * E_); // 16384x3072
    unsigned short* FF1  = QKV + (size_t)MROWS * 3072;           // 16384x2048
    unsigned short* ATTN = FF1 + (size_t)MROWS * 2048;           // 16384x1024
    unsigned short* EBF  = ATTN + (size_t)MROWS * E_;            // 16384x1024
    unsigned short* XBF  = EBF + (size_t)MROWS * E_;             // 16384x1024
    unsigned short* SBF  = XBF + (size_t)MROWS * E_;             // 8192x512
    unsigned short* ABF  = SBF + (size_t)B_ * S_ * SD_;          // 8192x256
    unsigned short* WBF  = ABF + (size_t)B_ * S_ * AD_;          // <= 3072x1024

    float* out = (float*)d_out;

    // ---- One-time bf16 conversion of the raw inputs ----
    {
        size_t ns = (size_t)B_ * S_ * SD_;
        cvt_kernel<<<(unsigned)((ns + 255) / 256), 256, 0, stream>>>(states, SBF, ns);
        size_t na = (size_t)B_ * S_ * AD_;
        cvt_kernel<<<(unsigned)((na + 255) / 256), 256, 0, stream>>>(actions, ABF, na);
    }

    // ---- Embeddings: es = states@Ws^T + bs ; ea = actions@Wa^T + ba ----
    launch_gemm(stream, SBF, SD_, Ws, WBF, bs_, TMP, false, E_,
                B_ * S_, E_, SD_, false);
    launch_gemm(stream, ABF, AD_, Wa, WBF, ba_, TMP + (size_t)B_ * S_ * E_,
                false, E_, B_ * S_, E_, AD_, false);
    {
        size_t n = (size_t)B_ * S_ * E_;
        interleave_kernel<<<(unsigned)((n + 255) / 256), 256, 0, stream>>>(
            TMP, TMP + n, X, XBF, EBF);
    }

    // ---- Encoder layers ----
    for (int l = 0; l < 4; ++l) {
        const int base = 9 + 12 * l;
        const float* Wqkv = (const float*)d_in[base + 0];
        const float* bqkv = (const float*)d_in[base + 1];
        const float* Wo   = (const float*)d_in[base + 2];
        const float* bo   = (const float*)d_in[base + 3];
        const float* W1   = (const float*)d_in[base + 4];
        const float* bff1 = (const float*)d_in[base + 5];
        const float* W2   = (const float*)d_in[base + 6];
        const float* bff2 = (const float*)d_in[base + 7];
        const float* g1   = (const float*)d_in[base + 8];
        const float* b1   = (const float*)d_in[base + 9];
        const float* g2   = (const float*)d_in[base + 10];
        const float* b2   = (const float*)d_in[base + 11];

        launch_gemm(stream, XBF, E_, Wqkv, WBF, bqkv, QKV, true, 3072,
                    MROWS, 3072, E_, false);
        attn_enc_kernel<<<dim3(T2_, H_), 32, 0, stream>>>(QKV, ATTN);
        launch_gemm(stream, ATTN, E_, Wo, WBF, bo, TMP, false, E_,
                    MROWS, E_, E_, false);
        ln_kernel<<<MROWS, 256, 0, stream>>>(X, TMP, g1, b1, X, XBF);
        launch_gemm(stream, XBF, E_, W1, WBF, bff1, FF1, true, F_,
                    MROWS, F_, E_, true);
        launch_gemm(stream, FF1, F_, W2, WBF, bff2, TMP, false, E_,
                    MROWS, E_, F_, false);
        ln_kernel<<<MROWS, 256, 0, stream>>>(X, TMP, g2, b2, X, XBF);
    }

    // ---- Action prediction head: x[:, -1, :] @ outW^T + outB -> (32, 256) ----
    launch_gemm(stream, XBF + (size_t)(T2_ - 1) * E_, (long)T2_ * E_,
                outW, WBF, outB, out, false, AD_, B_, AD_, E_, false);

    // ---- Forward-dynamics head ----
    launch_gemm(stream, EBF, E_, fWqkv, WBF, fbqkv, QKV, true, 3072,
                MROWS, 3072, E_, false);
    fdh_attn_kernel<<<dim3(S_ - 1, B_, H_), 32, 0, stream>>>(QKV, ATTN);
    const int Mfdh = (S_ - 1) * B_;  // 8160
    launch_gemm(stream, ATTN, E_, fWo, WBF, fbo, FF1, true, E_,
                Mfdh, E_, E_, false);
    float* o_dyn = out + (size_t)B_ * AD_;
    launch_gemm(stream, FF1, E_, fWout, WBF, fbout, o_dyn, false, SD_,
                Mfdh, SD_, E_, false);

    // ---- dyn_inv: zeros ----
    float* o_inv = o_dyn + (size_t)Mfdh * SD_;
    {
        size_t n = (size_t)Mfdh * AD_;
        fillz_kernel<<<(unsigned)((n + 255) / 256), 256, 0, stream>>>(o_inv, n);
    }

    // ---- Masked-action head ----
    launch_gemm(stream, EBF, E_, mWqkv, WBF, mbqkv, QKV, true, 3072,
                MROWS, 3072, E_, false);
    mah_attn_kernel<<<dim3(Ma, B_, H_), 32, 0, stream>>>(QKV, mia, umi, U, ATTN);
    const int Mmah = Ma * B_;
    launch_gemm(stream, ATTN, E_, mWo, WBF, mbo, FF1, true, E_,
                Mmah, E_, E_, false);
    launch_gemm(stream, FF1, E_, mWout, WBF, mbout, TMP, false, AD_,
                Mmah, AD_, E_, false);
    float* o_mask = o_inv + (size_t)Mfdh * AD_;
    {
        size_t n = (size_t)Mmah * AD_;
        scatter_masked_kernel<<<(unsigned)((n + 255) / 256), 256, 0, stream>>>(
            TMP, o_mask, Ma);
    }

    // ---- Trailing index outputs ----
    float* o_idx = o_mask + (size_t)B_ * Ma * AD_;
    idxcopy_kernel<<<1, 256, 0, stream>>>(mis, nMis, o_idx);
    idxcopy_kernel<<<1, 256, 0, stream>>>(mia, Ma, o_idx + nMis);
}
